// Attention_40097814675570
// MI455X (gfx1250) — matrix operand order repaired and checked by reading, not compiled, not run on hardware
//
#include <hip/hip_runtime.h>

// ---------------------------------------------------------------------------
// MI455X (gfx1250) fused attention:
//   bf16 WMMA + TDM (tensor_load_to_lds) K-staging + async V-staging
// ---------------------------------------------------------------------------

typedef __attribute__((ext_vector_type(16))) __bf16 bf16x16;
typedef __attribute__((ext_vector_type(8)))  float  f32x8;
typedef __attribute__((ext_vector_type(4)))  unsigned u32x4;
typedef __attribute__((ext_vector_type(8)))  unsigned u32x8;

#define L_SEQ  1024
#define HEADS  8
#define DHEAD  64
#define CDIM   512
#define QKV_LD 1536
#define BATCH  16

union FragBF {
    bf16x16 v;
    uint4   q[2];
};

__device__ __forceinline__ unsigned short f2bf(float f) {
    unsigned u = __float_as_uint(f);
    u += 0x7FFFu + ((u >> 16) & 1u);   // round-to-nearest-even
    return (unsigned short)(u >> 16);
}
__device__ __forceinline__ float bf2f(unsigned short h) {
    return __uint_as_float(((unsigned)h) << 16);
}

__device__ __forceinline__ f32x8 wmma_bf16(bf16x16 a, bf16x16 b, f32x8 c) {
    // (neg_a, A, neg_b, B, c_mod, C, reuse_a, reuse_b)
    return __builtin_amdgcn_wmma_f32_16x16x32_bf16(false, a, false, b,
                                                   (short)0, c, false, false);
}

// ---------------------------------------------------------------------------
// TDM: one-instruction DMA of a 64x64 bf16 tile (rows strided QKV_LD elems)
// from global memory into a contiguous row-major LDS buffer. Issued once per
// wave; tracked by TENSORcnt. D# per CDNA5 ISA ch.8 (2D, groups 0+1 only).
// ---------------------------------------------------------------------------
__device__ __forceinline__ void tdm_load_tile(const unsigned short* g,
                                              const unsigned short* lds) {
    const unsigned long long ga = (unsigned long long)(uintptr_t)g;
    u32x4 g0;
    g0[0] = 1u;                                    // count=1, user descriptor
    g0[1] = (unsigned)(uintptr_t)lds;              // lds_addr (bytes)
    g0[2] = (unsigned)(ga & 0xffffffffu);          // global_addr[31:0]
    g0[3] = (unsigned)((ga >> 32) & 0x01ffffffu)   // global_addr[56:32]
            | (2u << 30);                          // type = 2 ("image")
    u32x8 g1;
    g1[0] = 1u << 16;            // workgroup_mask=0, data_size=1 (2 bytes)
    g1[1] = 64u << 16;           // tensor_dim0[15:0]=64 in bits [63:48]
    g1[2] = (unsigned)L_SEQ << 16; // tensor_dim0 hi=0, tensor_dim1[15:0]=1024
    g1[3] = 64u << 16;           // tensor_dim1 hi=0, tile_dim0=64
    g1[4] = 64u;                 // tile_dim1=64, tile_dim2=0 (2D)
    g1[5] = (unsigned)QKV_LD;    // tensor_dim0_stride[31:0] = 1536 elements
    g1[6] = 0u;                  // stride hi, tensor_dim1_stride lo
    g1[7] = 0u;
    asm volatile("tensor_load_to_lds %0, %1" ::"s"(g0), "s"(g1) : "memory");
}

// Async copy of one 64x64 bf16 tile (8 KB) into row-major LDS.
// 128 threads x 16B x 4 rounds; tracked by ASYNCcnt (4 per thread).
__device__ __forceinline__ void async_tile_copy(
    const unsigned short* __restrict__ gbase, const unsigned short* lds_dst,
    int tid) {
    const unsigned ldsbase = (unsigned)(uintptr_t)lds_dst;  // low 32 = LDS off
#pragma unroll
    for (int rnd = 0; rnd < 4; ++rnd) {
        const int li   = rnd * 2048 + tid * 16;  // byte index within tile
        const int row  = li >> 7;                // /128 bytes per row
        const int colb = li & 127;
        const unsigned long long ga =
            (unsigned long long)(uintptr_t)(gbase + (size_t)row * QKV_LD) +
            (unsigned long long)colb;
        asm volatile("global_load_async_to_lds_b128 %0, %1, off" ::"v"(
                         ldsbase + (unsigned)li),
                     "v"(ga)
                     : "memory");
    }
}

// ---------------------------------------------------------------------------
// fp32 -> bf16 convert (vectorized, 4 elements / thread)
// ---------------------------------------------------------------------------
__global__ void cvt_f32_bf16_kernel(const float* __restrict__ in,
                                    unsigned short* __restrict__ out, int n4) {
    int i = blockIdx.x * blockDim.x + threadIdx.x;
    if (i >= n4) return;
    const float4 f = *(const float4*)(in + (size_t)i * 4);
    uint2 o;
    o.x = (unsigned)f2bf(f.x) | ((unsigned)f2bf(f.y) << 16);
    o.y = (unsigned)f2bf(f.z) | ((unsigned)f2bf(f.w) << 16);
    *(uint2*)(out + (size_t)i * 4) = o;
}

// ---------------------------------------------------------------------------
// Transpose + convert: W [K,N] fp32 -> WT [N,K] bf16 (B-fragment friendly)
// ---------------------------------------------------------------------------
__global__ void cvt_transpose_kernel(const float* __restrict__ W,
                                     unsigned short* __restrict__ WT,
                                     int K, int N) {
    int i = blockIdx.x * blockDim.x + threadIdx.x;
    if (i >= N * K) return;
    int n = i / K, k = i - n * K;
    WT[i] = f2bf(W[(size_t)k * N + n]);
}

// ---------------------------------------------------------------------------
// Relative position bias MLP: [L,L,1] -> leakyrelu -> [L,L,H], bf16 [H,L,L]
// ---------------------------------------------------------------------------
__global__ void bias_mlp_kernel(const float* __restrict__ W1,
                                const float* __restrict__ b1,
                                const float* __restrict__ W2,
                                unsigned short* __restrict__ biasb) {
    const int idx = blockIdx.x * blockDim.x + threadIdx.x; // over L*L
    const int i = idx >> 10, j = idx & (L_SEQ - 1);
    float tab = 0.f;
    if (i > 0 && j > 0) {
        const float rel = 8.0f * (float)(j - i) / 1022.0f;
        const float t = __log2f(fabsf(rel) + 1.0f) * (1.0f / 3.16992500144231f);
        tab = (rel > 0.f) ? t : ((rel < 0.f) ? -t : 0.f);
    }
    float acc[HEADS];
#pragma unroll
    for (int hh = 0; hh < HEADS; ++hh) acc[hh] = 0.f;
#pragma unroll 8
    for (int k = 0; k < 64; ++k) {
        float hv = tab * W1[k] + b1[k];
        hv = (hv > 0.f) ? hv : 0.2f * hv;   // LeakyReLU(0.2)
#pragma unroll
        for (int hh = 0; hh < HEADS; ++hh) acc[hh] += hv * W2[k * HEADS + hh];
    }
#pragma unroll
    for (int hh = 0; hh < HEADS; ++hh)
        biasb[(size_t)hh * (L_SEQ * L_SEQ) + idx] = f2bf(acc[hh]);
}

// ---------------------------------------------------------------------------
// bf16 WMMA GEMM:  C[M,N] = A[M,K] * BT[N,K]^T   (block 128x64, 8 waves 32x32)
// ---------------------------------------------------------------------------
template <int F32OUT>
__global__ __launch_bounds__(256) void gemm_bf16_kernel(
    const unsigned short* __restrict__ A,
    const unsigned short* __restrict__ BT,
    void* __restrict__ Cv, int M, int N, int K) {
    const int lane = threadIdx.x & 31;
    const int wid  = threadIdx.x >> 5;
    const int wm = wid & 3, wn = wid >> 2;
    const int m0 = blockIdx.y * 128 + wm * 32;
    const int n0 = blockIdx.x * 64 + wn * 32;
    const int arow = lane & 15;
    const int ksel = (lane < 16) ? 0 : 16;   // A-frag K-half select
    const int bsel = (lane < 16) ? 0 : 16;   // B-frag K-half select
    const int crow = (lane < 16) ? 0 : 8;    // C-frag row offset
    const f32x8 z8 = {0, 0, 0, 0, 0, 0, 0, 0};
    f32x8 acc[2][2];
    acc[0][0] = z8; acc[0][1] = z8; acc[1][0] = z8; acc[1][1] = z8;

    for (int k0 = 0; k0 < K; k0 += 32) {
        FragBF a[2], bfr[2];
#pragma unroll
        for (int i = 0; i < 2; ++i) {
            const unsigned short* ap = A + (size_t)(m0 + i * 16 + arow) * K + k0;
            a[i].q[0] = *(const uint4*)(ap + ksel);
            a[i].q[1] = *(const uint4*)(ap + ksel + 8);
            const unsigned short* bp = BT + (size_t)(n0 + i * 16 + arow) * K + k0 + bsel;
            bfr[i].q[0] = *(const uint4*)(bp);
            bfr[i].q[1] = *(const uint4*)(bp + 8);
        }
#pragma unroll
        for (int i = 0; i < 2; ++i)
#pragma unroll
            for (int j = 0; j < 2; ++j)
                acc[i][j] = wmma_bf16(a[i].v, bfr[j].v, acc[i][j]);
    }
#pragma unroll
    for (int i = 0; i < 2; ++i)
#pragma unroll
        for (int j = 0; j < 2; ++j)
#pragma unroll
            for (int r = 0; r < 8; ++r) {
                const size_t row = (size_t)(m0 + i * 16 + r + crow);
                const size_t col = (size_t)(n0 + j * 16 + (lane & 15));
                if (F32OUT)
                    ((float*)Cv)[row * N + col] = acc[i][j][r];
                else
                    ((unsigned short*)Cv)[row * N + col] = f2bf(acc[i][j][r]);
            }
}

// ---------------------------------------------------------------------------
// Fused flash attention: 4 waves, M=64 query tile, 64-key tiles.
// K tiles staged by the Tensor Data Mover (TENSORcnt), V tiles by async
// global->LDS loads (ASYNCcnt), both double-buffered. bf16 WMMA for QK^T and
// PV, fp32 online softmax, bias+mask fused. qkv layout [B,L,3,H,d] (row 1536).
// ---------------------------------------------------------------------------
__global__ __launch_bounds__(128) void flash_attn_kernel(
    const unsigned short* __restrict__ qkv,
    const unsigned short* __restrict__ biasb,   // bf16 [H,L,L]
    const unsigned char*  __restrict__ mask,    // bool [B, L-1]
    unsigned short* __restrict__ aout) {        // bf16 [B,L,C] (C = h*64+dd)
    __shared__ __align__(16) unsigned short lds_k[2][64 * 64];  // K tiles
    __shared__ __align__(16) unsigned short lds_v[2][64 * 64];  // V tiles raw
    __shared__ __align__(16) unsigned short lds_vt[64 * 72];    // V^T tile
    __shared__ __align__(16) unsigned short lds_p[4 * 16 * 64]; // P per wave

    const int tid  = threadIdx.x;
    const int lane = tid & 31;
    const int wave = tid >> 5;
    const int qt = blockIdx.x, h = blockIdx.y, b = blockIdx.z;
    const int qrow0 = qt * 64 + wave * 16;

    const int arow = lane & 15;
    const int ksel = (lane < 16) ? 0 : 16;
    const int bsel = (lane < 16) ? 0 : 16;
    const int crow = (lane < 16) ? 0 : 8;
    const f32x8 z8 = {0, 0, 0, 0, 0, 0, 0, 0};

    // Q fragments: resident for whole kernel (d=64 -> two K=32 chunks)
    FragBF qf[2];
    {
        const unsigned short* qb =
            qkv + (size_t)(b * L_SEQ + qrow0 + arow) * QKV_LD + h * DHEAD;
#pragma unroll
        for (int kc = 0; kc < 2; ++kc) {
            qf[kc].q[0] = *(const uint4*)(qb + kc * 32 + ksel);
            qf[kc].q[1] = *(const uint4*)(qb + kc * 32 + ksel + 8);
        }
    }

    f32x8 o[4];
#pragma unroll
    for (int j = 0; j < 4; ++j) o[j] = z8;
    float mstat[8], lstat[8];
#pragma unroll
    for (int r = 0; r < 8; ++r) { mstat[r] = -1e30f; lstat[r] = 0.f; }

    const unsigned short* biasp = biasb + (size_t)h * (L_SEQ * L_SEQ);
    const unsigned char*  maskp = mask + (size_t)b * (L_SEQ - 1);
    unsigned short* pmine = lds_p + wave * (16 * 64);

    // base of this (b,h)'s K / V row streams
    const unsigned short* kvb =
        qkv + (size_t)(b * L_SEQ) * QKV_LD + h * DHEAD;

    // prologue: stage tile 0 -- K via TDM (wave 0 only), V via async loads
    if (wave == 0) tdm_load_tile(kvb + CDIM, lds_k[0]);
    async_tile_copy(kvb + 2 * CDIM, lds_v[0], tid);

    for (int kt = 0; kt < L_SEQ / 64; ++kt) {
        const int kbase = kt * 64;
        const int cur = kt & 1, nxt = cur ^ 1;
        // software pipeline: issue next tile's transfers, wait for current.
        // TDM ops are in-order per wave; async loads complete in order too.
        if (kt < L_SEQ / 64 - 1) {
            const unsigned short* nb = kvb + (size_t)(kbase + 64) * QKV_LD;
            async_tile_copy(nb + 2 * CDIM, lds_v[nxt], tid);
            if (wave == 0) tdm_load_tile(nb + CDIM, lds_k[nxt]);
            asm volatile("s_wait_asynccnt 0x4" ::: "memory");
            __builtin_amdgcn_s_wait_tensorcnt(1);
            // prefetch next key tile's bias rows (global_prefetch_b8)
            __builtin_prefetch(
                biasp + (size_t)(qrow0 + (lane & 15)) * L_SEQ + kbase + 64, 0, 1);
        } else {
            asm volatile("s_wait_asynccnt 0x0" ::: "memory");
            __builtin_amdgcn_s_wait_tensorcnt(0);
        }
        __syncthreads();  // every wave's transfers for `cur` have landed

        // ---- S = Q K^T via WMMA, K fragments straight from LDS ----
        f32x8 s[4];
        const unsigned short* kl = lds_k[cur];
#pragma unroll
        for (int t = 0; t < 4; ++t) {
            const unsigned short* krow = kl + (t * 16 + arow) * 64;
            FragBF k0f, k1f;
            k0f.q[0] = *(const uint4*)(krow + bsel);
            k0f.q[1] = *(const uint4*)(krow + bsel + 8);
            k1f.q[0] = *(const uint4*)(krow + 32 + bsel);
            k1f.q[1] = *(const uint4*)(krow + 32 + bsel + 8);
            s[t] = wmma_bf16(qf[0].v, k0f.v, z8);
            s[t] = wmma_bf16(qf[1].v, k1f.v, s[t]);
        }
        // ---- scale (d^-1/2 = 0.125) + rel bias + key mask ----
#pragma unroll
        for (int t = 0; t < 4; ++t) {
            const int keycol = kbase + t * 16 + (lane & 15);
            float madd = 0.f;
            if (keycol > 0 && maskp[keycol - 1] == 0) madd = -1e30f;
#pragma unroll
            for (int r = 0; r < 8; ++r) {
                const int qr = qrow0 + r + crow;
                const float bv = bf2f(biasp[(size_t)qr * L_SEQ + keycol]);
                s[t][r] = s[t][r] * 0.125f + bv + madd;
            }
        }
        // ---- online softmax: row reductions across 16-lane halves ----
#pragma unroll
        for (int r = 0; r < 8; ++r) {
            float v = fmaxf(fmaxf(s[0][r], s[1][r]), fmaxf(s[2][r], s[3][r]));
            v = fmaxf(v, __shfl_xor(v, 1, 32));
            v = fmaxf(v, __shfl_xor(v, 2, 32));
            v = fmaxf(v, __shfl_xor(v, 4, 32));
            v = fmaxf(v, __shfl_xor(v, 8, 32));
            const float newm = fmaxf(mstat[r], v);
            const float corr = __expf(mstat[r] - newm);
            mstat[r] = newm;
            lstat[r] *= corr;
#pragma unroll
            for (int j = 0; j < 4; ++j) o[j][r] *= corr;
        }
#pragma unroll
        for (int t = 0; t < 4; ++t)
#pragma unroll
            for (int r = 0; r < 8; ++r) s[t][r] = __expf(s[t][r] - mstat[r]);
#pragma unroll
        for (int r = 0; r < 8; ++r) {
            float v = (s[0][r] + s[1][r]) + (s[2][r] + s[3][r]);
            v += __shfl_xor(v, 1, 32);
            v += __shfl_xor(v, 2, 32);
            v += __shfl_xor(v, 4, 32);
            v += __shfl_xor(v, 8, 32);
            lstat[r] += v;
        }

        __syncthreads();  // prior iteration's P/VT consumers are done
        // ---- write P (bf16, row-major 16x64) to this wave's LDS region ----
#pragma unroll
        for (int t = 0; t < 4; ++t)
#pragma unroll
            for (int r = 0; r < 8; ++r)
                pmine[(r + crow) * 64 + t * 16 + (lane & 15)] = f2bf(s[t][r]);
        // ---- transpose staged V tile: lds_v[cur][key][d] -> lds_vt[d][key] ----
        {
            const int vr = tid >> 1;         // key row 0..63
            const int ch = (tid & 1) * 32;   // d-column half
            const unsigned short* vrow = lds_v[cur] + vr * 64 + ch;
            union { uint4 q[4]; unsigned short u[32]; } vl;
#pragma unroll
            for (int qq = 0; qq < 4; ++qq)
                vl.q[qq] = *(const uint4*)(vrow + qq * 8);
#pragma unroll
            for (int c = 0; c < 32; ++c)
                lds_vt[(ch + c) * 72 + vr] = vl.u[c];
        }
        __syncthreads();
        // ---- O += P * V via WMMA from LDS ----
#pragma unroll
        for (int kc = 0; kc < 2; ++kc) {
            const int k0 = kc * 32;
            FragBF pa;
            pa.q[0] = *(const uint4*)(pmine + arow * 64 + k0 + ksel);
            pa.q[1] = *(const uint4*)(pmine + arow * 64 + k0 + ksel + 8);
#pragma unroll
            for (int j = 0; j < 4; ++j) {
                FragBF bv;
                const unsigned short* vtp =
                    lds_vt + (size_t)(j * 16 + (lane & 15)) * 72 + k0 + bsel;
                bv.q[0] = *(const uint4*)(vtp);
                bv.q[1] = *(const uint4*)(vtp + 8);
                o[j] = wmma_bf16(pa.v, bv.v, o[j]);
            }
        }
    }

    // ---- normalize and store bf16 [B, L, h*64+dd] ----
    unsigned short* op = aout + (size_t)(b * L_SEQ + qrow0) * CDIM + h * DHEAD;
#pragma unroll
    for (int j = 0; j < 4; ++j)
#pragma unroll
        for (int r = 0; r < 8; ++r) {
            const float v = o[j][r] / lstat[r];
            op[(size_t)(r + crow) * CDIM + j * 16 + (lane & 15)] = f2bf(v);
        }
}

// ---------------------------------------------------------------------------
// Host-side launch
// ---------------------------------------------------------------------------
extern "C" void kernel_launch(void* const* d_in, const int* in_sizes, int n_in,
                              void* d_out, int out_size, void* d_ws,
                              size_t ws_size, hipStream_t stream) {
    (void)in_sizes; (void)n_in; (void)out_size; (void)ws_size;
    const float*         x    = (const float*)d_in[0];
    const unsigned char* msk  = (const unsigned char*)d_in[1];
    const float*         Wqkv = (const float*)d_in[2];
    const float*         W1   = (const float*)d_in[3];
    const float*         b1   = (const float*)d_in[4];
    const float*         W2   = (const float*)d_in[5];
    const float*         Wout = (const float*)d_in[6];
    float*               out  = (float*)d_out;

    // workspace layout (bf16 elements), total ~103 MB
    unsigned short* ws16  = (unsigned short*)d_ws;
    unsigned short* xbf   = ws16;                  // 16384*512   =  8388608
    unsigned short* wqkvT = xbf + 8388608;         // 1536*512    =   786432
    unsigned short* woutT = wqkvT + 786432;        // 512*512     =   262144
    unsigned short* qkvb  = woutT + 262144;        // 16384*1536  = 25165824
    unsigned short* biasb = qkvb + 25165824;       // 8*1024*1024 =  8388608
    unsigned short* aoutb = biasb + 8388608;       // 16384*512   =  8388608

    // 1) convert x -> bf16
    cvt_f32_bf16_kernel<<<8388608 / (256 * 4), 256, 0, stream>>>(x, xbf,
                                                                 8388608 / 4);
    // 2) transpose-convert weights -> bf16 [N,K]
    cvt_transpose_kernel<<<786432 / 256, 256, 0, stream>>>(Wqkv, wqkvT,
                                                           CDIM, 3 * CDIM);
    cvt_transpose_kernel<<<262144 / 256, 256, 0, stream>>>(Wout, woutT,
                                                           CDIM, CDIM);
    // 3) relative position bias MLP -> bf16 [H,L,L]
    bias_mlp_kernel<<<(L_SEQ * L_SEQ) / 256, 256, 0, stream>>>(W1, b1, W2,
                                                               biasb);
    // 4) QKV GEMM: [16384,512] x [512,1536] -> bf16 [16384,1536]
    gemm_bf16_kernel<0><<<dim3(3 * CDIM / 64, BATCH * L_SEQ / 128), 256, 0,
                          stream>>>(xbf, wqkvT, qkvb, BATCH * L_SEQ, 3 * CDIM,
                                    CDIM);
    // 5) fused flash attention (TDM K tiles + async V tiles, double-buffered)
    flash_attn_kernel<<<dim3(L_SEQ / 64, HEADS, BATCH), 128, 0, stream>>>(
        qkvb, biasb, msk, aoutb);
    // 6) output projection: [16384,512] x [512,512] -> fp32 d_out
    gemm_bf16_kernel<1><<<dim3(CDIM / 64, BATCH * L_SEQ / 128), 256, 0,
                          stream>>>(aoutb, woutT, out, BATCH * L_SEQ, CDIM,
                                    CDIM);
}
